// DeformableConvBlock_48137993453666
// MI455X (gfx1250) — matrix operand order repaired and checked by reading, not compiled, not run on hardware
//
#include <hip/hip_runtime.h>

// CDNA5 / gfx1250 deformable conv: both contractions as implicit GEMMs on
// v_wmma_f32_16x16x32_f16 (f16 A/B, f32 accumulate), wave32.
// A-tile staging via the Tensor Data Mover (tensor_load_to_lds + TENSORcnt).

typedef __attribute__((ext_vector_type(16))) _Float16 v16h;
typedef __attribute__((ext_vector_type(8)))  float    v8f;
typedef __attribute__((ext_vector_type(4)))  unsigned u32x4;
typedef __attribute__((ext_vector_type(8)))  int      i32x8;
typedef __attribute__((ext_vector_type(4)))  int      i32x4;

#if defined(__gfx1250__) && __has_builtin(__builtin_amdgcn_tensor_load_to_lds) && \
    __has_builtin(__builtin_amdgcn_s_wait_tensorcnt)
#define HAVE_TDM 1
#else
#define HAVE_TDM 0
#endif

#define BATCH 4
#define CIN   64
#define HH    128
#define WW    128
#define COUT  64
#define OFFC  18
#define CK    576      // K dimension = 9 taps * 64 channels
#define KSTEPS 18      // 576 / 32
#define LDA   40       // LDS row pitch (halves): 80B = 64B data + 16B pad, 16B-aligned
#define LDB   40

// masked read inside one channel plane; &127 keeps the address in-bounds even
// when the tap is outside the image (value is then zeroed, matching reference)
__device__ __forceinline__ float sample_plane(const float* __restrict__ xc, int y, int xx) {
  bool ok = ((unsigned)y < (unsigned)HH) && ((unsigned)xx < (unsigned)WW);
  float v = xc[(y & (HH - 1)) * WW + (xx & (WW - 1))];
  return ok ? v : 0.0f;
}

#if HAVE_TDM
// TDM: DMA one A chunk (rows x 32 halves, global row stride 576 halves) into
// LDS with 16B padding after every 64B row -> LDS pitch 80B (= LDA halves).
// D# bitfields per CDNA5 ISA ch.8 (group0: count/lds/global/type=2,
// group1: data_size=2B, pad_interval=16 DW, pad_amount=4 DW, dims/strides).
// 6-arg builtin form (clang-23 lane): trailing i32x8 = unused SGPR group.
__device__ __forceinline__ void tdm_load_a_chunk(const _Float16* __restrict__ gsrc,
                                                 unsigned lds_off, int rows) {
  unsigned long long ga = (unsigned long long)(uintptr_t)gsrc;
  u32x4 g0;
  g0[0] = 1u;                                        // count=1, user descriptor
  g0[1] = lds_off;                                   // lds_addr (bytes)
  g0[2] = (unsigned)(ga & 0xFFFFFFFFu);              // global_addr[31:0]
  g0[3] = (unsigned)((ga >> 32) & 0x01FFFFFFu)       // global_addr[56:32]
        | (2u << 30);                                // type = 2 ("image")
  i32x8 g1;
  g1[0] = (int)((1u << 16)                           // data_size = 2 bytes
              | (1u << 20)                           // pad_enable
              | (3u << 22)                           // pad_interval: 16 DWORDs (64B)
              | (3u << 25));                         // pad_amount: 4 DWORDs (16B)
  g1[1] = (int)(576u << 16);                         // tensor_dim0 = 576 (low 16)
  g1[2] = (int)((unsigned)rows << 16);               // dim0 hi=0 | tensor_dim1 = rows
  g1[3] = (int)(32u << 16);                          // dim1 hi=0 | tile_dim0 = 32
  g1[4] = rows;                                      // tile_dim1 = rows, tile_dim2 = 0
  g1[5] = 576;                                       // tensor_dim0_stride[31:0]
  g1[6] = 0;                                         // stride0 hi | stride1 lo
  g1[7] = 0;                                         // tensor_dim1_stride hi
  i32x4 z4 = {0, 0, 0, 0};
  i32x8 z8 = {0, 0, 0, 0, 0, 0, 0, 0};
  __builtin_amdgcn_tensor_load_to_lds(g0, g1, z4, z4, z8, 0);
}
#endif

// ---------------------------------------------------------------------------
// prep: build f16 A-matrices with K reordered as k = tap*64 + c.
// A1: 32 x 576 (rows 18..31 zero-padded) from off_w(18,64,3,3)
// A2: 64 x 576 from weight(64,64,3,3)
// ---------------------------------------------------------------------------
__global__ void prep_kernel(const float* __restrict__ off_w,
                            const float* __restrict__ weight,
                            _Float16* __restrict__ A1,
                            _Float16* __restrict__ A2) {
  int i = blockIdx.x * blockDim.x + threadIdx.x;
  if (i < 32 * CK) {
    int m = i / CK, k = i % CK;
    int t = k >> 6, c = k & 63;
    float v = (m < OFFC) ? off_w[((size_t)m * CIN + c) * 9 + t] : 0.0f;
    A1[i] = (_Float16)v;
  } else if (i < 32 * CK + COUT * CK) {
    int j = i - 32 * CK;
    int m = j / CK, k = j % CK;
    int t = k >> 6, c = k & 63;
    A2[j] = (_Float16)weight[((size_t)m * CIN + c) * 9 + t];
  }
}

// ---------------------------------------------------------------------------
// Fragment loaders (CDNA5 16-bit WMMA VGPR striping, wave32).
// A 16x32: lane m=0..15, halves 0..7 -> K = 2*kbase..2*kbase+7,
//          halves 8..15 -> K = 2*kbase+8..; kbase = (lane>>4)*8
// B 32x16: lane n, halves 0..15 -> K = (lane>>4)*16 + e (contiguous)
// ---------------------------------------------------------------------------
__device__ __forceinline__ v16h load_a_frag(const _Float16* __restrict__ As,
                                            int mt, int lrow, int kbase) {
  v16h a;
  const _Float16* arow = &As[(mt * 16 + lrow) * LDA];
#pragma unroll
  for (int e = 0; e < 8; ++e) {
    a[e]     = arow[2 * kbase + e];
    a[8 + e] = arow[2 * kbase + 8 + e];
  }
  return a;
}

__device__ __forceinline__ v16h load_b_frag(const _Float16* __restrict__ Bs,
                                            int nt, int lrow, int half_sel) {
  v16h b;
  const _Float16* brow = &Bs[(nt * 16 + lrow) * LDB + half_sel * 16];
#pragma unroll
  for (int e = 0; e < 16; ++e) b[e] = brow[e];
  return b;
}

// ---------------------------------------------------------------------------
// Stage 1: offset = conv3x3(x, off_w) + off_b, via WMMA implicit GEMM.
// Block: 128 threads = 4 waves = 2 M-tiles x 2 N-tiles over a 32-pixel strip.
// ---------------------------------------------------------------------------
__global__ __launch_bounds__(128) void offset_conv_kernel(
    const float* __restrict__ x, const _Float16* __restrict__ A1,
    const float* __restrict__ off_b, float* __restrict__ offset_out) {
  __shared__ __align__(16) _Float16 As[32 * LDA];
  __shared__ __align__(16) _Float16 Bs[32 * LDB];

  const int tid  = threadIdx.x;
  const int lane = tid & 31;
  const int wave = tid >> 5;          // 0..3
  const int mt = wave & 1, nt = wave >> 1;
  const int half_sel = lane >> 4;
  const int lrow  = lane & 15;
  const int kbase = half_sel * 8;

  const int tile = blockIdx.x;        // 2048 tiles of 32 pixels
  const int b   = tile >> 9;          // 512 tiles per batch image
  const int rem = tile & 511;
  const int h   = rem >> 2;           // 4 tiles per row
  const int w0  = (rem & 3) * 32;

  const int bn   = tid & 31;          // pixel-in-tile for staging
  const int cgrp = tid >> 5;          // 0..3

  v8f acc = {};

  for (int ks = 0; ks < KSTEPS; ++ks) {
    // stage A chunk: 32 rows x 32 halves
#if HAVE_TDM
    if (wave == 0)
      tdm_load_a_chunk(A1 + ks * 32, (unsigned)(uintptr_t)(void*)&As[0], 32);
#else
    {
      int base = tid * 8;             // 128*8 == 1024 exactly
      int m = base >> 5, kk = base & 31;
      *reinterpret_cast<float4*>(&As[m * LDA + kk]) =
          *reinterpret_cast<const float4*>(A1 + (size_t)m * CK + ks * 32 + kk);
    }
#endif
    // stage B chunk: shifted 3x3 patch values for one tap, 32 channels
    {
      int t  = ks >> 1;
      int c0 = (ks & 1) * 32;
      int yy = h + (t / 3) - 1;
      int xx = w0 + bn + (t % 3) - 1;
      const float* xb = x + (size_t)b * CIN * (HH * WW);
#pragma unroll
      for (int j = 0; j < 8; ++j) {
        int cl = cgrp * 8 + j;
        float v = sample_plane(xb + (size_t)(c0 + cl) * (HH * WW), yy, xx);
        Bs[bn * LDB + cl] = (_Float16)v;
      }
    }
#if HAVE_TDM
    if (wave == 0) __builtin_amdgcn_s_wait_tensorcnt(0);
#endif
    __syncthreads();
    v16h a  = load_a_frag(As, mt, lrow, kbase);
    v16h bf = load_b_frag(Bs, nt, lrow, half_sel);
    acc = __builtin_amdgcn_wmma_f32_16x16x32_f16(false, a, false, bf,
                                                 (short)0, acc, false, false);
    __syncthreads();
  }

#pragma unroll
  for (int r = 0; r < 8; ++r) {
    int m = mt * 16 + half_sel * 8 + r;
    int n = nt * 16 + lrow;
    if (m < OFFC) {
      offset_out[(((size_t)b * OFFC + m) * HH + h) * WW + (w0 + n)] = acc[r] + off_b[m];
    }
  }
}

// ---------------------------------------------------------------------------
// Stage 2: bilinear-sample + einsum('bckhw,ock->bohw') via WMMA.
// Block: 256 threads = 8 waves = 4 M-tiles x 2 N-tiles over a 32-pixel strip.
// Each bilinear sample is gathered once into LDS and reused by 64 out rows.
// ---------------------------------------------------------------------------
__global__ __launch_bounds__(256) void deform_gemm_kernel(
    const float* __restrict__ x, const _Float16* __restrict__ A2,
    const float* __restrict__ offset, const float* __restrict__ bias,
    float* __restrict__ out) {
  __shared__ __align__(16) _Float16 As[64 * LDA];
  __shared__ __align__(16) _Float16 Bs[32 * LDB];
  __shared__ int   sp_y0[9 * 32];
  __shared__ int   sp_x0[9 * 32];
  __shared__ float sp_w[4][9 * 32];

  const int tid  = threadIdx.x;
  const int lane = tid & 31;
  const int wave = tid >> 5;          // 0..7
  const int mt = wave & 3, nt = wave >> 2;
  const int half_sel = lane >> 4;
  const int lrow  = lane & 15;
  const int kbase = half_sel * 8;

  const int tile = blockIdx.x;
  const int b   = tile >> 9;
  const int rem = tile & 511;
  const int h   = rem >> 2;
  const int w0  = (rem & 3) * 32;

  // Precompute bilinear params for 9 taps x 32 pixels
  for (int i = tid; i < 9 * 32; i += 256) {
    int t = i >> 5, n = i & 31;
    float oy = offset[(((size_t)b * OFFC + 2 * t) * HH + h) * WW + (w0 + n)];
    float ox = offset[(((size_t)b * OFFC + 2 * t + 1) * HH + h) * WW + (w0 + n)];
    float sy = oy + (float)(h + (t / 3) - 1);
    float sx = ox + (float)(w0 + n + (t % 3) - 1);
    float fy = floorf(sy), fx = floorf(sx);
    float wy1 = sy - fy, wx1 = sx - fx;
    float wy0 = 1.0f - wy1, wx0 = 1.0f - wx1;
    sp_y0[i] = (int)fy;
    sp_x0[i] = (int)fx;
    sp_w[0][i] = wy0 * wx0;
    sp_w[1][i] = wy0 * wx1;
    sp_w[2][i] = wy1 * wx0;
    sp_w[3][i] = wy1 * wx1;
  }
  __syncthreads();

  const int bn   = tid & 31;
  const int cgrp = tid >> 5;          // 0..7

  v8f acc = {};

  for (int ks = 0; ks < KSTEPS; ++ks) {
    // stage A chunk: 64 rows x 32 halves
#if HAVE_TDM
    if (wave == 0)
      tdm_load_a_chunk(A2 + ks * 32, (unsigned)(uintptr_t)(void*)&As[0], 64);
#else
    {
      int base = tid * 8;             // 256*8 == 2048 exactly
      int m = base >> 5, kk = base & 31;
      *reinterpret_cast<float4*>(&As[m * LDA + kk]) =
          *reinterpret_cast<const float4*>(A2 + (size_t)m * CK + ks * 32 + kk);
    }
#endif
    // stage B chunk: bilinear samples for one tap, 32 channels x 32 pixels
    {
      int t  = ks >> 1;
      int c0 = (ks & 1) * 32;
      int pi = t * 32 + bn;
      int y0 = sp_y0[pi], x0 = sp_x0[pi];
      float w00 = sp_w[0][pi], w01 = sp_w[1][pi];
      float w10 = sp_w[2][pi], w11 = sp_w[3][pi];
      const float* xb = x + (size_t)b * CIN * (HH * WW);
#pragma unroll
      for (int j = 0; j < 4; ++j) {
        int cl = cgrp + 8 * j;
        const float* xc = xb + (size_t)(c0 + cl) * (HH * WW);
        float v = w00 * sample_plane(xc, y0,     x0)
                + w01 * sample_plane(xc, y0,     x0 + 1)
                + w10 * sample_plane(xc, y0 + 1, x0)
                + w11 * sample_plane(xc, y0 + 1, x0 + 1);
        Bs[bn * LDB + cl] = (_Float16)v;
      }
    }
#if HAVE_TDM
    if (wave == 0) __builtin_amdgcn_s_wait_tensorcnt(0);
#endif
    __syncthreads();
    v16h a  = load_a_frag(As, mt, lrow, kbase);
    v16h bf = load_b_frag(Bs, nt, lrow, half_sel);
    acc = __builtin_amdgcn_wmma_f32_16x16x32_f16(false, a, false, bf,
                                                 (short)0, acc, false, false);
    __syncthreads();
  }

#pragma unroll
  for (int r = 0; r < 8; ++r) {
    int m = mt * 16 + half_sel * 8 + r;
    int n = nt * 16 + lrow;
    out[(((size_t)b * COUT + m) * HH + h) * WW + (w0 + n)] = acc[r] + bias[m];
  }
}

// ---------------------------------------------------------------------------
extern "C" void kernel_launch(void* const* d_in, const int* in_sizes, int n_in,
                              void* d_out, int out_size, void* d_ws, size_t ws_size,
                              hipStream_t stream) {
  const float* x      = (const float*)d_in[0];   // (4,64,128,128)
  const float* off_w  = (const float*)d_in[1];   // (18,64,3,3)
  const float* off_b  = (const float*)d_in[2];   // (18,)
  const float* weight = (const float*)d_in[3];   // (64,64,3,3)
  const float* bias   = (const float*)d_in[4];   // (64,)
  float* out = (float*)d_out;                    // (4,64,128,128)

  char* ws = (char*)d_ws;
  float*    offset_buf = (float*)ws;                          // 4*18*128*128*4 = 4718592 B
  _Float16* A1 = (_Float16*)(ws + 4718592);                   // 32*576*2 = 36864 B
  _Float16* A2 = (_Float16*)(ws + 4718592 + 36864);           // 64*576*2 = 73728 B

  const int prep_n = 32 * CK + COUT * CK;                     // 55296 elements
  prep_kernel<<<(prep_n + 255) / 256, 256, 0, stream>>>(off_w, weight, A1, A2);
  offset_conv_kernel<<<2048, 128, 0, stream>>>(x, A1, off_b, offset_buf);
  deform_gemm_kernel<<<2048, 256, 0, stream>>>(x, A2, offset_buf, bias, out);
}
